// MENet_88665304858932
// MI455X (gfx1250) — matrix operand order repaired and checked by reading, not compiled, not run on hardware
//
#include <hip/hip_runtime.h>
#include <cstdint>

// ---------------------------------------------------------------------------
// Types for CDNA5 WMMA (wave32): D(16x16 f32) = A(16x32 f16) x B(32x16 f16) + C
// ---------------------------------------------------------------------------
typedef _Float16 v16h __attribute__((ext_vector_type(16)));
typedef _Float16 v8h  __attribute__((ext_vector_type(8)));
typedef float    v8f  __attribute__((ext_vector_type(8)));

__device__ __forceinline__ v8f wmma_f16f32(v16h a, v16h b, v8f c) {
    return __builtin_amdgcn_wmma_f32_16x16x32_f16(
        /*neg_a=*/false, a, /*neg_b=*/false, b,
        /*c_mod=*/(short)0, c, /*reuse_a=*/false, /*reuse_b=*/false);
}

// A-fragment (16x32 f16, MxK) from a row-major [*, pitch] f16 matrix.
// Lane l: m = l&15, half = l>>4.
//   a[j]   = A[m][kbase + 16*half + j]      j in 0..7
//   a[8+j] = A[m][kbase + 16*half + 8 + j]
__device__ __forceinline__ v16h load_a_frag(const _Float16* __restrict__ base,
                                            int row0, int pitch, int kbase, int lane) {
    int m = lane & 15, half = lane >> 4;
    const _Float16* p = base + (size_t)(row0 + m) * pitch + kbase + half * 16;
    v8h lo = *(const v8h*)p;
    v8h hi = *(const v8h*)(p + 8);
    v16h a;
#pragma unroll
    for (int i = 0; i < 8; ++i) { a[i] = lo[i]; a[8 + i] = hi[i]; }
    return a;
}

// B-fragment (32x16 f16, KxN) where B = W^T and W is row-major [N][K] f16.
// Lane l: n = l&15, half = l>>4.  b[j] = W[row_n+n][kbase + 16*half + j]
__device__ __forceinline__ v16h load_b_frag(const _Float16* __restrict__ w,
                                            int row_n, int K, int kbase, int lane) {
    int n = lane & 15, half = lane >> 4;
    return *(const v16h*)(w + (size_t)(row_n + n) * K + kbase + half * 16);
}

// ---------------------------------------------------------------------------
// Prep kernels
// ---------------------------------------------------------------------------
__global__ void f32_to_f16_kernel(const float* __restrict__ src,
                                  _Float16* __restrict__ dst, int n) {
    int i = blockIdx.x * blockDim.x + threadIdx.x;
    if (i < n) dst[i] = (_Float16)src[i];
}

// fc3: pad [40,256] -> [48,256] f16 (zero rows), pad bias 40 -> 48
__global__ void prep_fc3_kernel(const float* __restrict__ w40,
                                const float* __restrict__ b40,
                                _Float16* __restrict__ w48,
                                float* __restrict__ b48) {
    int i = blockIdx.x * blockDim.x + threadIdx.x;
    if (i < 48 * 256) {
        int r = i >> 8;
        w48[i] = (r < 40) ? (_Float16)w40[i] : (_Float16)0.0f;
    }
    if (i < 48) b48[i] = (i < 40) ? b40[i] : 0.0f;
}

// s = g/sqrt(1+eps); bias = has_fcb ? fc_b*s + bn_b : bn_b
__global__ void prep_bn_kernel(const float* __restrict__ g,
                               const float* __restrict__ fcb,
                               const float* __restrict__ bnb,
                               float* __restrict__ s, float* __restrict__ bias,
                               int n, int has_fcb) {
    int i = blockIdx.x * blockDim.x + threadIdx.x;
    if (i < n) {
        float sc = g[i] * rsqrtf(1.0f + 1e-5f);
        s[i] = sc;
        bias[i] = has_fcb ? (fcb[i] * sc + bnb[i]) : bnb[i];
    }
}

// row-normalize memory_w [16,64]
__global__ void mnorm_kernel(const float* __restrict__ mw, float* __restrict__ mn) {
    __shared__ float inv[16];
    int t = threadIdx.x;
    if (t < 16) {
        float ss = 0.0f;
        for (int c = 0; c < 64; ++c) { float v = mw[t * 64 + c]; ss += v * v; }
        inv[t] = 1.0f / fmaxf(sqrtf(ss), 1e-12f);
    }
    __syncthreads();
    for (int i = t; i < 16 * 64; i += blockDim.x) mn[i] = mw[i] * inv[i >> 6];
}

// ---------------------------------------------------------------------------
// Pooling kernels (HBM-bound): wave-per-row max reductions
// ---------------------------------------------------------------------------
__global__ __launch_bounds__(256) void l3max_kernel(const float* __restrict__ l3,
                                                    float* __restrict__ l3m,
                                                    _Float16* __restrict__ xf16) {
    int row  = blockIdx.x * 8 + (threadIdx.x >> 5);   // 512*1024 rows of 128
    int lane = threadIdx.x & 31;
    const float* p = l3 + (size_t)row * 128;
    float v = -3.4e38f;
#pragma unroll
    for (int i = 0; i < 4; ++i) v = fmaxf(v, p[lane + 32 * i]);
#pragma unroll
    for (int off = 16; off > 0; off >>= 1) v = fmaxf(v, __shfl_xor(v, off, 32));
    if (lane == 0) {
        l3m[row] = v;
        int b = row >> 10, j = row & 1023;
        xf16[(size_t)b * 1280 + 256 + j] = (_Float16)v;   // x tail = max(l3,2)
    }
}

__global__ __launch_bounds__(256) void x2max_kernel(const float* __restrict__ x2,
                                                    const float* __restrict__ l3m,
                                                    _Float16* __restrict__ x2f16) {
    int row  = blockIdx.x * 8 + (threadIdx.x >> 5);   // 512*1024 rows of 256
    int lane = threadIdx.x & 31;
    const float* p = x2 + (size_t)row * 256;
    float v = -3.4e38f;
#pragma unroll
    for (int i = 0; i < 8; ++i) v = fmaxf(v, p[lane + 32 * i]);
#pragma unroll
    for (int off = 16; off > 0; off >>= 1) v = fmaxf(v, __shfl_xor(v, off, 32));
    if (lane == 0) {
        int b = row >> 10, j = row & 1023;
        x2f16[(size_t)b * 1536 + 512 + j] = (_Float16)(v + l3m[row]); // x2 tail
    }
}

// ---------------------------------------------------------------------------
// Branch kernel: memory addressing + 2-layer MLP (WMMA) + max-pool over N=32
// One block (8 waves) per batch element.
// ---------------------------------------------------------------------------
__global__ __launch_bounds__(256) void branch_kernel(
    const float* __restrict__ feat_g,   // [B,64,32]
    const float* __restrict__ memw,     // [16,64]
    const float* __restrict__ mnorm_g,  // [16,64]
    const _Float16* __restrict__ w1, const float* __restrict__ s1,
    const float* __restrict__ bb1, int C1,
    const _Float16* __restrict__ w2, const float* __restrict__ s2,
    const float* __restrict__ bb2, int C2,
    _Float16* __restrict__ outp, int out_pitch) {
    __shared__ float featS[64 * 32];
    __shared__ float mnS[16 * 64];
    __shared__ float mwS[16 * 64];
    __shared__ float colinv[32];
    __shared__ float logitS[32 * 16];
    __shared__ float aS[32 * 16];
    __shared__ __align__(16) _Float16 A0[32 * 64];
    __shared__ __align__(16) _Float16 H1[32 * 256];

    int b = blockIdx.x;
    int t = threadIdx.x;

    for (int i = t; i < 64 * 32; i += 256) featS[i] = feat_g[(size_t)b * 2048 + i];
    for (int i = t; i < 16 * 64; i += 256) { mnS[i] = mnorm_g[i]; mwS[i] = memw[i]; }
    __syncthreads();

    // per-point inverse L2 norm over channels (columns of featS)
    if (t < 32) {
        float ss = 0.0f;
        for (int c = 0; c < 64; ++c) { float v = featS[c * 32 + t]; ss += v * v; }
        colinv[t] = 1.0f / fmaxf(sqrtf(ss), 1e-12f);
    }
    __syncthreads();

    // logits[n][s] = (feat[:,n]/||.||) . mnorm[s]
    for (int i = t; i < 512; i += 256) {
        int n = i >> 4, s = i & 15;
        float d = 0.0f;
        for (int c = 0; c < 64; ++c) d += featS[c * 32 + n] * mnS[s * 64 + c];
        logitS[n * 16 + s] = d * colinv[n];
    }
    __syncthreads();

    // softmax over 16 slots per point
    if (t < 32) {
        float mx = -3.4e38f;
        for (int s = 0; s < 16; ++s) mx = fmaxf(mx, logitS[t * 16 + s]);
        float e[16], sum = 0.0f;
        for (int s = 0; s < 16; ++s) { e[s] = __expf(logitS[t * 16 + s] - mx); sum += e[s]; }
        float inv = 1.0f / sum;
        for (int s = 0; s < 16; ++s) aS[t * 16 + s] = e[s] * inv;
    }
    __syncthreads();

    // addressed features m[n][c] = a[n] . memw[:,c]   -> f16 A0 [32][64]
    for (int i = t; i < 32 * 64; i += 256) {
        int n = i >> 6, c = i & 63;
        float acc = 0.0f;
        for (int s = 0; s < 16; ++s) acc += aS[n * 16 + s] * mwS[s * 64 + c];
        A0[n * 64 + c] = (_Float16)acc;
    }
    __syncthreads();

    int wave = t >> 5, lane = t & 31;

    // layer 1: [32,64] x W1^T -> H1 [32,C1], BN+ReLU
    int nt1 = C1 >> 4;
    for (int nt = wave; nt < nt1; nt += 8) {
        int n = nt * 16 + (lane & 15);
        float sc = s1[n], bi = bb1[n];
        int half = lane >> 4;
        for (int mt = 0; mt < 2; ++mt) {
            v8f acc = {};
#pragma unroll
            for (int kt = 0; kt < 2; ++kt) {
                v16h a = load_a_frag(A0, mt * 16, 64, kt * 32, lane);
                v16h bf = load_b_frag(w1, nt * 16, 64, kt * 32, lane);
                acc = wmma_f16f32(a, bf, acc);
            }
#pragma unroll
            for (int j = 0; j < 8; ++j) {
                float v = fmaxf(acc[j] * sc + bi, 0.0f);
                H1[(mt * 16 + half * 8 + j) * C1 + n] = (_Float16)v;
            }
        }
    }
    __syncthreads();

    // layer 2: [32,C1] x W2^T -> [32,C2], BN+ReLU, fused max-pool over 32 pts
    int nt2 = C2 >> 4;
    int nk2 = C1 >> 5;
    for (int nt = wave; nt < nt2; nt += 8) {
        int n = nt * 16 + (lane & 15);
        float sc = s2[n], bi = bb2[n];
        float mx = -3.4e38f;
        for (int mt = 0; mt < 2; ++mt) {
            v8f acc = {};
            for (int kt = 0; kt < nk2; ++kt) {
                v16h a = load_a_frag(H1, mt * 16, C1, kt * 32, lane);
                v16h bf = load_b_frag(w2, nt * 16, C1, kt * 32, lane);
                acc = wmma_f16f32(a, bf, acc);
            }
#pragma unroll
            for (int j = 0; j < 8; ++j)
                mx = fmaxf(mx, fmaxf(acc[j] * sc + bi, 0.0f));
        }
        mx = fmaxf(mx, __shfl_xor(mx, 16, 32));   // combine halves -> all 32 pts
        if (lane < 16) outp[(size_t)b * out_pitch + n] = (_Float16)mx;
    }
}

// ---------------------------------------------------------------------------
// Head GEMM: out[512,N] = act(BN(X[512,K] @ W^T)); wave per 16x16 tile
// ---------------------------------------------------------------------------
__global__ __launch_bounds__(128) void head_gemm_kernel(
    const _Float16* __restrict__ X, int K,
    const _Float16* __restrict__ W,
    const float* __restrict__ s, const float* __restrict__ bias,
    int N, int relu,
    _Float16* __restrict__ out_h, float* __restrict__ out_f) {
    int wave = threadIdx.x >> 5, lane = threadIdx.x & 31;
    int tiles_n = N >> 4;
    int gw = blockIdx.x * 4 + wave;
    int mt = gw / tiles_n, nt = gw % tiles_n;

    v8f acc = {};
    int nk = K >> 5;
    for (int kt = 0; kt < nk; ++kt) {
        v16h a  = load_a_frag(X, mt * 16, K, kt * 32, lane);
        v16h bf = load_b_frag(W, nt * 16, K, kt * 32, lane);
        acc = wmma_f16f32(a, bf, acc);
    }
    int n = nt * 16 + (lane & 15);
    int half = lane >> 4;
    float sc = s ? s[n] : 1.0f;
    float bi = bias[n];
#pragma unroll
    for (int j = 0; j < 8; ++j) {
        int m = mt * 16 + half * 8 + j;
        float v = acc[j] * sc + bi;
        if (relu) v = fmaxf(v, 0.0f);
        if (out_h) out_h[(size_t)m * N + n] = (_Float16)v;
        else       out_f[(size_t)m * N + n] = v;
    }
}

// log_softmax over 40 classes (pitch 48), wave per row
__global__ __launch_bounds__(256) void logsoftmax_kernel(const float* __restrict__ lg,
                                                         float* __restrict__ out) {
    int row  = blockIdx.x * 8 + (threadIdx.x >> 5);
    int lane = threadIdx.x & 31;
    const float* p = lg + (size_t)row * 48;
    float x0 = p[lane];
    float x1 = (lane < 8) ? p[lane + 32] : -3.4e38f;
    float mx = fmaxf(x0, x1);
#pragma unroll
    for (int off = 16; off > 0; off >>= 1) mx = fmaxf(mx, __shfl_xor(mx, off, 32));
    float e = __expf(x0 - mx) + ((lane < 8) ? __expf(x1 - mx) : 0.0f);
#pragma unroll
    for (int off = 16; off > 0; off >>= 1) e += __shfl_xor(e, off, 32);
    float lse = mx + logf(e);
    out[(size_t)row * 40 + lane] = x0 - lse;
    if (lane < 8) out[(size_t)row * 40 + 32 + lane] = x1 - lse;
}

// ---------------------------------------------------------------------------
// Launcher
// ---------------------------------------------------------------------------
extern "C" void kernel_launch(void* const* d_in, const int* in_sizes, int n_in,
                              void* d_out, int out_size, void* d_ws, size_t ws_size,
                              hipStream_t stream) {
    const float* mem_f1   = (const float*)d_in[0];
    const float* mem_f2   = (const float*)d_in[1];
    const float* l3       = (const float*)d_in[2];
    const float* x2p      = (const float*)d_in[3];
    const float* memw     = (const float*)d_in[4];
    const float* m1w1 = (const float*)d_in[5];  const float* m1g1 = (const float*)d_in[6];  const float* m1b1 = (const float*)d_in[7];
    const float* m1w2 = (const float*)d_in[8];  const float* m1g2 = (const float*)d_in[9];  const float* m1b2 = (const float*)d_in[10];
    const float* m2w1 = (const float*)d_in[11]; const float* m2g1 = (const float*)d_in[12]; const float* m2b1 = (const float*)d_in[13];
    const float* m2w2 = (const float*)d_in[14]; const float* m2g2 = (const float*)d_in[15]; const float* m2b2 = (const float*)d_in[16];
    const float* fc1w = (const float*)d_in[17]; const float* fc1b = (const float*)d_in[18];
    const float* bn1g = (const float*)d_in[19]; const float* bn1b = (const float*)d_in[20];
    const float* fc2w = (const float*)d_in[21]; const float* fc2b = (const float*)d_in[22];
    const float* bn2g = (const float*)d_in[23]; const float* bn2b = (const float*)d_in[24];
    const float* fc3w = (const float*)d_in[25]; const float* fc3b = (const float*)d_in[26];
    const float* f12w = (const float*)d_in[27]; const float* f12b = (const float*)d_in[28];
    const float* b12g = (const float*)d_in[29]; const float* b12b = (const float*)d_in[30];
    const float* f22w = (const float*)d_in[31]; const float* f22b = (const float*)d_in[32];
    const float* b22g = (const float*)d_in[33]; const float* b22b = (const float*)d_in[34];
    const float* f32w = (const float*)d_in[35]; const float* f32b = (const float*)d_in[36];
    float* out = (float*)d_out;

    // workspace allocator (256B aligned)
    size_t off = 0;
    auto alloc = [&](size_t bytes) -> char* {
        char* p = (char*)d_ws + off;
        off += (bytes + 255) & ~(size_t)255;
        return p;
    };
    float*     mnorm  = (float*)alloc(16 * 64 * 4);
    float*     l3m    = (float*)alloc((size_t)512 * 1024 * 4);
    _Float16*  xf16   = (_Float16*)alloc((size_t)512 * 1280 * 2);
    _Float16*  x2f16  = (_Float16*)alloc((size_t)512 * 1536 * 2);
    _Float16*  w11    = (_Float16*)alloc(128 * 64 * 2);
    _Float16*  w12    = (_Float16*)alloc(256 * 128 * 2);
    _Float16*  w21    = (_Float16*)alloc(256 * 64 * 2);
    _Float16*  w22    = (_Float16*)alloc(512 * 256 * 2);
    _Float16*  wf1    = (_Float16*)alloc((size_t)512 * 1280 * 2);
    _Float16*  wf2    = (_Float16*)alloc(256 * 512 * 2);
    _Float16*  wf3    = (_Float16*)alloc(48 * 256 * 2);
    _Float16*  wf1b   = (_Float16*)alloc((size_t)512 * 1536 * 2);
    _Float16*  wf2b   = (_Float16*)alloc(256 * 512 * 2);
    _Float16*  wf3b   = (_Float16*)alloc(48 * 256 * 2);
    float* s11 = (float*)alloc(128 * 4); float* c11 = (float*)alloc(128 * 4);
    float* s12 = (float*)alloc(256 * 4); float* c12 = (float*)alloc(256 * 4);
    float* s21 = (float*)alloc(256 * 4); float* c21 = (float*)alloc(256 * 4);
    float* s22 = (float*)alloc(512 * 4); float* c22 = (float*)alloc(512 * 4);
    float* sh1 = (float*)alloc(512 * 4); float* ch1 = (float*)alloc(512 * 4);
    float* sh2 = (float*)alloc(256 * 4); float* ch2 = (float*)alloc(256 * 4);
    float* ch3 = (float*)alloc(48 * 4);
    float* sh1b = (float*)alloc(512 * 4); float* ch1b = (float*)alloc(512 * 4);
    float* sh2b = (float*)alloc(256 * 4); float* ch2b = (float*)alloc(256 * 4);
    float* ch3b = (float*)alloc(48 * 4);
    _Float16* h1     = (_Float16*)alloc((size_t)512 * 512 * 2);
    _Float16* h2     = (_Float16*)alloc((size_t)512 * 256 * 2);
    float*    logits = (float*)alloc((size_t)512 * 48 * 4);
    (void)ws_size; (void)n_in; (void)in_sizes; (void)out_size;

    auto cvt = [&](const float* src, _Float16* dst, int n) {
        f32_to_f16_kernel<<<(n + 255) / 256, 256, 0, stream>>>(src, dst, n);
    };

    // ---- prep ----
    mnorm_kernel<<<1, 256, 0, stream>>>(memw, mnorm);
    cvt(m1w1, w11, 128 * 64);   cvt(m1w2, w12, 256 * 128);
    cvt(m2w1, w21, 256 * 64);   cvt(m2w2, w22, 512 * 256);
    cvt(fc1w, wf1, 512 * 1280); cvt(fc2w, wf2, 256 * 512);
    cvt(f12w, wf1b, 512 * 1536); cvt(f22w, wf2b, 256 * 512);
    prep_fc3_kernel<<<(48 * 256 + 255) / 256, 256, 0, stream>>>(fc3w, fc3b, wf3, ch3);
    prep_fc3_kernel<<<(48 * 256 + 255) / 256, 256, 0, stream>>>(f32w, f32b, wf3b, ch3b);
    prep_bn_kernel<<<1, 256, 0, stream>>>(m1g1, nullptr, m1b1, s11, c11, 128, 0);
    prep_bn_kernel<<<1, 256, 0, stream>>>(m1g2, nullptr, m1b2, s12, c12, 256, 0);
    prep_bn_kernel<<<1, 256, 0, stream>>>(m2g1, nullptr, m2b1, s21, c21, 256, 0);
    prep_bn_kernel<<<2, 256, 0, stream>>>(m2g2, nullptr, m2b2, s22, c22, 512, 0);
    prep_bn_kernel<<<2, 256, 0, stream>>>(bn1g, fc1b, bn1b, sh1, ch1, 512, 1);
    prep_bn_kernel<<<1, 256, 0, stream>>>(bn2g, fc2b, bn2b, sh2, ch2, 256, 1);
    prep_bn_kernel<<<2, 256, 0, stream>>>(b12g, f12b, b12b, sh1b, ch1b, 512, 1);
    prep_bn_kernel<<<1, 256, 0, stream>>>(b22g, f22b, b22b, sh2b, ch2b, 256, 1);

    // ---- HBM-bound pooling (dominant cost) ----
    l3max_kernel<<<512 * 1024 / 8, 256, 0, stream>>>(l3, l3m, xf16);
    x2max_kernel<<<512 * 1024 / 8, 256, 0, stream>>>(x2p, l3m, x2f16);

    // ---- memory-addressing branches (WMMA) ----
    branch_kernel<<<512, 256, 0, stream>>>(mem_f1, memw, mnorm,
                                           w11, s11, c11, 128,
                                           w12, s12, c12, 256,
                                           xf16, 1280);
    branch_kernel<<<512, 256, 0, stream>>>(mem_f2, memw, mnorm,
                                           w21, s21, c21, 256,
                                           w22, s22, c22, 512,
                                           x2f16, 1536);

    // ---- head 1: 1280 -> 512 -> 256 -> 40 ----
    head_gemm_kernel<<<32 * 32 / 4, 128, 0, stream>>>(xf16, 1280, wf1, sh1, ch1, 512, 1, h1, nullptr);
    head_gemm_kernel<<<32 * 16 / 4, 128, 0, stream>>>(h1, 512, wf2, sh2, ch2, 256, 1, h2, nullptr);
    head_gemm_kernel<<<32 * 3 / 4, 128, 0, stream>>>(h2, 256, wf3, nullptr, ch3, 48, 0, nullptr, logits);
    logsoftmax_kernel<<<512 / 8, 256, 0, stream>>>(logits, out);

    // ---- head 2: 1536 -> 512 -> 256 -> 40 ----
    head_gemm_kernel<<<32 * 32 / 4, 128, 0, stream>>>(x2f16, 1536, wf1b, sh1b, ch1b, 512, 1, h1, nullptr);
    head_gemm_kernel<<<32 * 16 / 4, 128, 0, stream>>>(h1, 512, wf2b, sh2b, ch2b, 256, 1, h2, nullptr);
    head_gemm_kernel<<<32 * 3 / 4, 128, 0, stream>>>(h2, 256, wf3b, nullptr, ch3b, 48, 0, nullptr, logits);
    logsoftmax_kernel<<<512 / 8, 256, 0, stream>>>(logits, out + (size_t)512 * 40);
}
